// LocalReadLayer_51256139710508
// MI455X (gfx1250) — matrix operand order repaired and checked by reading, not compiled, not run on hardware
//
#include <hip/hip_runtime.h>
#include <hip/hip_bf16.h>

typedef __attribute__((ext_vector_type(16))) _Float16 v16h;
typedef __attribute__((ext_vector_type(8)))  _Float16 v8h;
typedef __attribute__((ext_vector_type(8)))  float    v8f;
typedef __attribute__((ext_vector_type(4)))  unsigned int v4u;
typedef __attribute__((ext_vector_type(8)))  int v8i;
typedef __attribute__((ext_vector_type(4)))  int v4i;

#define BDIM   2
#define KV     256
#define HIMG   256
#define WIMG   256
#define QDIM   1024
#define HEADS  8
#define WIN    32
#define NPX    64                     // 8x8 pixels per latent block
#define MROWS  (BDIM * WIN * WIN)     // 2048 latent rows total

__device__ __forceinline__ v8f wmma16x16x32(v16h a, v16h b, v8f c) {
  return __builtin_amdgcn_wmma_f32_16x16x32_f16(
      /*neg_a=*/false, a, /*neg_b=*/false, b,
      /*c_mod=*/(short)0, c, /*reuse_a=*/false, /*reuse_b=*/false);
}

// Low 32 bits of a generic pointer to LDS == wave-relative LDS byte address
// (ISA 10.2: LDS aperture maps by truncation).
__device__ __forceinline__ unsigned int lds_off(const void* p) {
  return (unsigned int)(unsigned long long)(uintptr_t)p;
}

// CDNA5 async copy global -> LDS, 16B per lane, tracked with ASYNCcnt.
__device__ __forceinline__ void async_b128(unsigned int ldsDst, unsigned int vOff,
                                           const void* base) {
  asm volatile("global_load_async_to_lds_b128 %0, %1, %2"
               :
               : "v"(ldsDst), "v"(vOff),
                 "s"((unsigned long long)(uintptr_t)base)
               : "memory");
}
__device__ __forceinline__ void wait_async0() {
  asm volatile("s_wait_asynccnt 0x0" ::: "memory");
}

// A fragment (16x32 f16) from a row-major [16+][ld] matrix, K window at ks.
// lane<16 -> M=lane, K = {ks..ks+7} U {ks+16..ks+23};
// lane>=16 -> M=lane-16, K = {ks+8..ks+15} U {ks+24..ks+31}.
__device__ __forceinline__ v16h fragA_rm(const _Float16* base, int ld, int lane, int ks) {
  int r = lane & 15;
  int o = (lane < 16) ? 0 : 8;
  const _Float16* p = base + (size_t)r * ld + ks + o;
  v16h f;
  ((v8h*)&f)[0] = *(const v8h*)(p);
  ((v8h*)&f)[1] = *(const v8h*)(p + 16);
  return f;
}

// B fragment (32x16 f16) with the matrix stored [N][K] row-major (A @ W^T).
__device__ __forceinline__ v16h fragB_nk(const _Float16* base, int ld, int lane, int ks) {
  int n = lane & 15;
  int o = (lane < 16) ? 0 : 8;
  const _Float16* p = base + (size_t)n * ld + ks + o;
  v16h f;
  ((v8h*)&f)[0] = *(const v8h*)(p);
  ((v8h*)&f)[1] = *(const v8h*)(p + 16);
  return f;
}

// ---------------------------------------------------------------------------
// Kernel 1: fused logits -> softmax -> pixel aggregation for one 8x8 block.
// The 256ch x 8 x 8 patch is fetched by the Tensor Data Mover as one 3D tile
// (tile_dim={8,8,256}, strides {256,65536}), then converted f32->f16 in LDS.
// b_log dropped (softmax shift-invariant per head); b_val folded into the
// value GEMM (attention weights sum to 1).
// ---------------------------------------------------------------------------
__global__ __launch_bounds__(256) void k_local_attend(
    const float* __restrict__ xhr, const float* __restrict__ w_log,
    _Float16* __restrict__ agg16)
{
  __shared__ float s_raw[KV * NPX];    // TDM destination: [ch][py][px] f32, 64KB
  __shared__ v8h s_xpk[NPX][KV / 8];   // X as [px][ch]   (B operand for logits)
  __shared__ v8h s_xkp[KV][NPX / 8];   // X as [ch][px]   (B operand for agg)
  __shared__ v8h s_wl[16][KV / 8];     // w_log padded to 16 heads
  __shared__ float s_lg[16][NPX];      // logits
  __shared__ v8h s_at[16][NPX / 8];    // softmax weights, rows 8..15 zero
  _Float16* Xpk = (_Float16*)s_xpk;
  _Float16* Xkp = (_Float16*)s_xkp;
  _Float16* Wl  = (_Float16*)s_wl;
  _Float16* At  = (_Float16*)s_at;

  const int tid  = threadIdx.x;
  const int wave = tid >> 5, lane = tid & 31;
  const int blk  = blockIdx.x;              // 0..1023 = i*32+j
  const int b    = blockIdx.y;              // batch
  const int y0   = (blk >> 5) * 8, x0 = (blk & 31) * 8;

  // ---- TDM: one tensor_load_to_lds per workgroup (wave 0 issues & waits) ----
  if (wave == 0) {
    const float* gptr = xhr + (size_t)b * KV * HIMG * WIMG
                            + (size_t)y0 * WIMG + x0;       // tile start
    unsigned long long ga = (unsigned long long)(uintptr_t)gptr;
    // D# group0: count=1, lds_addr, 57b global addr, type=2
    v4u g0 = { 1u, lds_off(s_raw), (unsigned int)ga,
               (unsigned int)(ga >> 32) | (2u << 30) };
    // D# group1: data_size=4B(code 2); tensor_dim0=256, tensor_dim1=256;
    // tile_dim0=8, tile_dim1=8, tile_dim2=256;
    // tensor_dim0_stride=256 (y), tensor_dim1_stride=65536 (channel plane)
    v8i g1 = { (int)0x20000,
               (int)(256u << 16),          // tensor_dim0 lo16 @ bits 63:48
               (int)(256u << 16),          // dim0 hi16=0 | tensor_dim1 lo16
               (int)(8u << 16),            // dim1 hi16=0 | tile_dim0=8
               (int)(8u | (256u << 16)),   // tile_dim1=8 | tile_dim2=256
               256,                        // tensor_dim0_stride lo32
               0,                          // stride0 hi16 | stride1 lo16 (=0)
               1 };                        // tensor_dim1_stride hi32 (65536>>16)
    // D# group2: tensor_dim2=256 (OOB bound on channels), rest unused
    v4i g2 = { 256, 1, 0, 0 };
    v4i g3 = { 0, 0, 0, 0 };
#if defined(__clang_major__) && (__clang_major__ >= 23)
    v8i gpad = { 0, 0, 0, 0, 0, 0, 0, 0 };
    __builtin_amdgcn_tensor_load_to_lds(g0, g1, g2, g3, gpad, 0);
#else
    __builtin_amdgcn_tensor_load_to_lds(g0, g1, g2, g3, 0);
#endif
    __builtin_amdgcn_s_wait_tensorcnt(0);
  }
  for (int e = tid; e < 16 * KV; e += 256) {
    int r = e >> 8, k = e & 255;
    Wl[r * KV + k] = (r < HEADS) ? (_Float16)w_log[r * KV + k] : (_Float16)0.0f;
  }
  for (int e = tid; e < 8 * NPX; e += 256)
    At[(8 + (e >> 6)) * NPX + (e & 63)] = (_Float16)0.0f;
  __syncthreads();

  // f32 -> f16 conversion into both major orders (raw index = ch*64 + px).
  for (int e = tid; e < KV * NPX; e += 256) {
    int k = e >> 6, p = e & 63;
    _Float16 h = (_Float16)s_raw[e];
    Xpk[p * KV + k] = h;
    Xkp[k * NPX + p] = h;
  }
  __syncthreads();

  // Logits: [16 heads x 64 px] = Wl[16,256] @ X[256,64].  4 waves, 1 px-tile each.
  if (wave < 4) {
    v8f acc = {};
    int p0 = wave * 16;
    for (int ks = 0; ks < KV; ks += 32) {
      v16h a  = fragA_rm(Wl, KV, lane, ks);
      v16h bf = fragB_nk(Xpk + (size_t)p0 * KV, KV, lane, ks);
      acc = wmma16x16x32(a, bf, acc);
    }
    int col = p0 + (lane & 15);
    int rb  = (lane < 16) ? 0 : 8;
#pragma unroll
    for (int g = 0; g < 8; ++g) s_lg[rb + g][col] = acc[g];
  }
  __syncthreads();

  // Softmax over 64 pixels: one wave per head.
  {
    int head = wave;
    float v0 = s_lg[head][lane], v1 = s_lg[head][lane + 32];
    float mx = fmaxf(v0, v1);
#pragma unroll
    for (int s = 16; s > 0; s >>= 1) mx = fmaxf(mx, __shfl_xor(mx, s, 32));
    float e0 = __expf(v0 - mx), e1 = __expf(v1 - mx);
    float sm = e0 + e1;
#pragma unroll
    for (int s = 16; s > 0; s >>= 1) sm += __shfl_xor(sm, s, 32);
    float inv = 1.0f / sm;
    At[head * NPX + lane]      = (_Float16)(e0 * inv);
    At[head * NPX + lane + 32] = (_Float16)(e1 * inv);
  }
  __syncthreads();

  // Aggregation: Agg[16 heads x 256 ch] = At[16,64] @ X[64,256].
  size_t obase = ((size_t)b * (WIN * WIN) + blk) * (HEADS * KV);
#pragma unroll
  for (int t = 0; t < 2; ++t) {
    int n0 = (wave * 2 + t) * 16;
    v8f acc = {};
#pragma unroll
    for (int ks = 0; ks < NPX; ks += 32) {
      v16h a  = fragA_rm(At, NPX, lane, ks);
      v16h bf = fragB_nk(Xkp + (size_t)n0 * NPX, NPX, lane, ks);
      acc = wmma16x16x32(a, bf, acc);
    }
    if (lane < 16) {                      // heads 0..7 live in lanes 0..15
      int ch = n0 + lane;
#pragma unroll
      for (int g = 0; g < 8; ++g)
        agg16[obase + (size_t)g * KV + ch] = (_Float16)acc[g];
    }
  }
}

// ---------------------------------------------------------------------------
// Generic WMMA GEMM: C[m,n] = epilogue( A[m,:] . W[n,:] + bias[n] ).
// Workgroup = 8 waves, tile 128M x 128N.  B tile staged via CDNA5 async
// global->LDS copies (double buffered, ASYNCcnt-fenced); A operand streamed
// from global with prefetch.  8 WMMA per wave per 32-K step.
// MODE 0: f32 out;  MODE 1: f32 out + residual;  MODE 2: exact GELU -> f16 out.
// grid.z batches independent GEMMs (per-head value projection).
// ---------------------------------------------------------------------------
template<int MODE>
__global__ __launch_bounds__(256) void k_gemm(
    const _Float16* __restrict__ A, int lda, long zA,
    const _Float16* __restrict__ W, int ldw, long zW,
    const float* __restrict__ bias, long zBias,
    const float* __restrict__ resid,
    float* __restrict__ Cf, _Float16* __restrict__ Ch, int ldc, long zC,
    int K)
{
  __shared__ v8h s_b[2][128][4];         // [buf][n][k/8], 32-K window, 16KB
  const int tid  = threadIdx.x;
  const int wave = tid >> 5, lane = tid & 31;
  const int n0 = blockIdx.x * 128;
  const int m0 = blockIdx.y * 128;
  const int z  = blockIdx.z;
  A    += (size_t)z * zA;
  W    += (size_t)z * zW;
  bias += (size_t)z * zBias;
  const long cOff = (long)z * zC;

  // Async staging pattern: 256 threads x 32B = 8KB per K window.
  const int ln = tid >> 1;               // 0..127: row of B tile
  const int lc = (tid & 1) * 16;         // half-row K chunk (16 halves)
  const unsigned int rowByte =
      (unsigned int)(((size_t)(n0 + ln) * ldw + lc) * 2);

  auto stage = [&](int bufIdx, int ks) {
    unsigned int vo = rowByte + (unsigned int)(ks * 2);
    unsigned int d  = lds_off((_Float16*)&s_b[bufIdx][ln][0] + lc);
    async_b128(d,      vo,      W);
    async_b128(d + 16, vo + 16, W);
  };

  stage(0, 0);
  wait_async0();
  __syncthreads();

  const _Float16* arow = A + (size_t)(m0 + wave * 16) * lda;
  const _Float16* apre = arow + (size_t)(lane & 15) * lda;   // prefetch cursor
  v8f acc[8] = {};
  int buf = 0;
  for (int ks = 0; ks < K; ks += 32) {
    if (ks + 32 < K) stage(buf ^ 1, ks + 32);          // async B prefetch
    if (ks + 64 < K) __builtin_prefetch(apre + ks + 64, 0, 0);
    v16h a = fragA_rm(arow, lda, lane, ks);
    const _Float16* bb = (const _Float16*)&s_b[buf][0][0];
#pragma unroll
    for (int j = 0; j < 8; ++j) {
      v16h bf = fragB_nk(bb + (size_t)(j * 16) * 32, 32, lane, 0);
      acc[j] = wmma16x16x32(a, bf, acc[j]);
    }
    wait_async0();                                     // B(buf^1) landed
    __syncthreads();
    buf ^= 1;
  }

  const int rbase = m0 + wave * 16 + ((lane < 16) ? 0 : 8);
#pragma unroll
  for (int j = 0; j < 8; ++j) {
    int col  = n0 + j * 16 + (lane & 15);
    float bv = bias[col];
#pragma unroll
    for (int g = 0; g < 8; ++g) {
      int row = rbase + g;
      float v = acc[j][g] + bv;
      size_t idx = (size_t)row * ldc + cOff + col;
      if (MODE == 0) {
        Cf[idx] = v;
      } else if (MODE == 1) {
        Cf[idx] = resid[idx] + v;
      } else {
        float ge = 0.5f * v * (1.0f + erff(v * 0.7071067811865476f));
        Ch[idx] = (_Float16)ge;
      }
    }
  }
}

// ---------------------------------------------------------------------------
// Row LayerNorm (D=1024), f32 in -> f16 out.  One workgroup per row.
// ---------------------------------------------------------------------------
__global__ __launch_bounds__(256) void k_layernorm(
    const float* __restrict__ X, const float* __restrict__ gam,
    const float* __restrict__ bet, _Float16* __restrict__ Y, int D)
{
  __shared__ float s_red[16];
  const int row = blockIdx.x;
  const float* x = X + (size_t)row * D;
  float s = 0.0f, s2 = 0.0f;
  for (int i = threadIdx.x; i < D; i += 256) { float v = x[i]; s += v; s2 += v * v; }
#pragma unroll
  for (int o = 16; o > 0; o >>= 1) { s += __shfl_xor(s, o, 32); s2 += __shfl_xor(s2, o, 32); }
  int wave = threadIdx.x >> 5, lane = threadIdx.x & 31;
  if (lane == 0) { s_red[wave] = s; s_red[8 + wave] = s2; }
  __syncthreads();
  if (wave == 0) {
    float a = (lane < 8) ? s_red[lane] : 0.0f;
    float b = (lane < 8) ? s_red[8 + lane] : 0.0f;
#pragma unroll
    for (int o = 4; o > 0; o >>= 1) { a += __shfl_xor(a, o, 32); b += __shfl_xor(b, o, 32); }
    if (lane == 0) {
      float mu  = a / (float)D;
      float var = b / (float)D - mu * mu;
      s_red[0] = mu;
      s_red[1] = rsqrtf(var + 1e-5f);
    }
  }
  __syncthreads();
  float mu = s_red[0], rstd = s_red[1];
  _Float16* y = Y + (size_t)row * D;
  for (int i = threadIdx.x; i < D; i += 256)
    y[i] = (_Float16)((x[i] - mu) * rstd * gam[i] + bet[i]);
}

__global__ void k_cvt(const float* __restrict__ src, _Float16* __restrict__ dst, int n) {
  int i = blockIdx.x * 256 + threadIdx.x;
  if (i < n) dst[i] = (_Float16)src[i];
}

// ---------------------------------------------------------------------------
extern "C" void kernel_launch(void* const* d_in, const int* in_sizes, int n_in,
                              void* d_out, int out_size, void* d_ws, size_t ws_size,
                              hipStream_t stream)
{
  (void)in_sizes; (void)n_in; (void)out_size; (void)ws_size;
  const float* xlr    = (const float*)d_in[0];
  const float* xhr    = (const float*)d_in[1];
  const float* w_log  = (const float*)d_in[2];
  /* d_in[3] = b_log: constant per head over the softmax axis -> no-op */
  const float* w_val  = (const float*)d_in[4];
  const float* b_val  = (const float*)d_in[5];
  const float* g1     = (const float*)d_in[6];
  const float* be1    = (const float*)d_in[7];
  const float* g2     = (const float*)d_in[8];
  const float* be2    = (const float*)d_in[9];
  const float* w_proj = (const float*)d_in[10];
  const float* b_proj = (const float*)d_in[11];
  const float* w_mlp1 = (const float*)d_in[12];
  const float* b_mlp1 = (const float*)d_in[13];
  const float* w_mlp2 = (const float*)d_in[14];
  const float* b_mlp2 = (const float*)d_in[15];
  float* out = (float*)d_out;

  char* ws = (char*)d_ws;
  size_t off = 0;
  auto take = [&](size_t bytes) -> char* {
    char* p = ws + off;
    off += (bytes + 255) & ~(size_t)255;
    return p;
  };
  _Float16* agg16   = (_Float16*)take((size_t)MROWS * HEADS * KV * 2);
  _Float16* wval16  = (_Float16*)take((size_t)QDIM * KV * 2);
  _Float16* wproj16 = (_Float16*)take((size_t)QDIM * QDIM * 2);
  _Float16* wm116   = (_Float16*)take((size_t)QDIM * QDIM * 2);
  _Float16* wm216   = (_Float16*)take((size_t)QDIM * QDIM * 2);
  float*    readb   = (float*)take((size_t)MROWS * QDIM * 4);
  _Float16* ln116   = (_Float16*)take((size_t)MROWS * QDIM * 2);
  float*    xlr1    = (float*)take((size_t)MROWS * QDIM * 4);
  _Float16* ln216   = (_Float16*)take((size_t)MROWS * QDIM * 2);
  _Float16* h16     = (_Float16*)take((size_t)MROWS * QDIM * 2);

  // Weight down-conversion (13 MB total, negligible vs 134 MB xhr pass).
  k_cvt<<<(QDIM * KV + 255) / 256, 256, 0, stream>>>(w_val, wval16, QDIM * KV);
  k_cvt<<<(QDIM * QDIM + 255) / 256, 256, 0, stream>>>(w_proj, wproj16, QDIM * QDIM);
  k_cvt<<<(QDIM * QDIM + 255) / 256, 256, 0, stream>>>(w_mlp1, wm116, QDIM * QDIM);
  k_cvt<<<(QDIM * QDIM + 255) / 256, 256, 0, stream>>>(w_mlp2, wm216, QDIM * QDIM);

  // 1) fused logits + softmax + pixel aggregation (single TDM pass over xhr).
  k_local_attend<<<dim3(WIN * WIN, BDIM), 256, 0, stream>>>(xhr, w_log, agg16);

  // 2) read = agg @ W_val_h^T + b_val   (per-head GEMM via grid.z)
  k_gemm<0><<<dim3(1, MROWS / 128, HEADS), 256, 0, stream>>>(
      agg16, HEADS * KV, (long)KV,
      wval16, KV, (long)(128 * KV),
      b_val, (long)128,
      nullptr, readb, nullptr, QDIM, (long)128, KV);

  // 3) xlr1 = xlr + LN(read) @ W_proj^T + b_proj
  k_layernorm<<<MROWS, 256, 0, stream>>>(readb, g1, be1, ln116, QDIM);
  k_gemm<1><<<dim3(QDIM / 128, MROWS / 128, 1), 256, 0, stream>>>(
      ln116, QDIM, 0L, wproj16, QDIM, 0L, b_proj, 0L,
      xlr, xlr1, nullptr, QDIM, 0L, QDIM);

  // 4) h = gelu(LN(xlr1) @ W_mlp1^T + b_mlp1)  (exact erf GELU)
  k_layernorm<<<MROWS, 256, 0, stream>>>(xlr1, g2, be2, ln216, QDIM);
  k_gemm<2><<<dim3(QDIM / 128, MROWS / 128, 1), 256, 0, stream>>>(
      ln216, QDIM, 0L, wm116, QDIM, 0L, b_mlp1, 0L,
      nullptr, nullptr, h16, QDIM, 0L, QDIM);

  // 5) out = xlr1 + h @ W_mlp2^T + b_mlp2
  k_gemm<1><<<dim3(QDIM / 128, MROWS / 128, 1), 256, 0, stream>>>(
      h16, QDIM, 0L, wm216, QDIM, 0L, b_mlp2, 0L,
      xlr1, out, nullptr, QDIM, 0L, QDIM);
}
